// GraphDecoder_gate_74388833566987
// MI455X (gfx1250) — matrix operand order repaired and checked by reading, not compiled, not run on hardware
//
#include <hip/hip_runtime.h>
#include <hip/hip_bf16.h>

// ---------------------------------------------------------------------------
// Problem constants (from reference)
// ---------------------------------------------------------------------------
#define NN    200000   // nodes
#define GG    8192     // graphs
#define DIN   256      // input dim
#define HH    512      // hidden dim

typedef __attribute__((ext_vector_type(16))) __bf16 bf16x16;
typedef __attribute__((ext_vector_type(8)))  __bf16 bf16x8;
typedef __attribute__((ext_vector_type(8)))  float  fx8;

// fp32 -> bf16 (round to nearest even), as raw bits
__device__ __forceinline__ unsigned short f2bfbits(float f) {
    unsigned int u = __builtin_bit_cast(unsigned int, f);
    unsigned int r = u + 0x7fffu + ((u >> 16) & 1u);
    return (unsigned short)(r >> 16);
}
__device__ __forceinline__ float bf2f(unsigned short h) {
    unsigned int u = ((unsigned int)h) << 16;
    return __builtin_bit_cast(float, u);
}

// ---------------------------------------------------------------------------
// Pack fp32 K x C weight matrix into CDNA5 WMMA B-operand order (bf16).
// Per 32(K) x 16(N) tile: dword index ((kt*CT + nt)*32 + lane)*8 + v holds
// bf16 pair ( W[kt*32 + 2v + 16*(lane>>4)][nt*16 + (lane&15)],
//             W[kt*32 + 2v + 16*(lane>>4) + 1][same col] ).
// ---------------------------------------------------------------------------
__global__ void pack_b_kernel(const float* __restrict__ W,
                              unsigned int* __restrict__ out, int K, int C) {
    int id = blockIdx.x * blockDim.x + threadIdx.x;
    int total = (K * C) >> 1;
    if (id >= total) return;
    int v    = id & 7;
    int L    = (id >> 3) & 31;
    int tile = id >> 8;
    int CT   = C >> 4;
    int nt   = tile % CT;
    int kt   = tile / CT;
    int kh   = L >> 4;
    int n    = L & 15;
    int k0   = kt * 32 + 2 * v + 16 * kh;
    unsigned int lo = f2bfbits(W[(size_t)k0 * C + nt * 16 + n]);
    unsigned int hi = f2bfbits(W[(size_t)(k0 + 1) * C + nt * 16 + n]);
    out[id] = (hi << 16) | lo;
}

// fp32 -> bf16 row-major bulk convert (4 elements / thread)
__global__ void cvt_bf16_kernel(const float4* __restrict__ in,
                                uint2* __restrict__ out, int n4) {
    int id = blockIdx.x * blockDim.x + threadIdx.x;
    if (id >= n4) return;
    float4 f = in[id];
    uint2 o;
    o.x = (unsigned int)f2bfbits(f.x) | ((unsigned int)f2bfbits(f.y) << 16);
    o.y = (unsigned int)f2bfbits(f.z) | ((unsigned int)f2bfbits(f.w) << 16);
    out[id] = o;
}

// ---------------------------------------------------------------------------
// Fused WMMA GEMM:  acc(16xC per wave) = X(16xK bf16) @ W(KxC bf16) + b1
//   MODE 0: store acc as bf16 rows into Ybf          (h = x@W + b)
//   MODE 1: Yvec[row] = dot(silu(acc_row), w2) + b2  (gate / head scalar)
// 8 waves per block, each wave owns one 16-row M tile; A resident in VGPRs;
// B chunks (NTCHUNK column tiles, all K) staged into 64KB LDS.
// Inner product pipelined: 2 rotating B buffers + 2 independent accumulators
// so consecutive v_wmma are not RAW-dependent and ds_load latency overlaps.
// ---------------------------------------------------------------------------
template <int KT, int CT, int MODE, int NTCHUNK>
__global__ __launch_bounds__(256) void mlp_kernel(
    const unsigned short* __restrict__ X,    // M x (KT*32) bf16 row-major
    const unsigned int*   __restrict__ Wp,   // packed B operand
    const float* __restrict__ b1,            // C
    const float* __restrict__ w2,            // C   (MODE 1)
    const float* __restrict__ b2,            // [1] (MODE 1)
    unsigned short* __restrict__ Ybf,        // M x C bf16 (MODE 0)
    float* __restrict__ Yvec,                // M floats   (MODE 1)
    int mtiles) {
    constexpr int Kdim = KT * 32;
    constexpr int C    = CT * 16;
    __shared__ alignas(32) unsigned int sW[KT * NTCHUNK * 256];  // 64 KB

    const int tid  = threadIdx.x;
    const int lane = tid & 31;
    const int wave = tid >> 5;
    const int mt   = blockIdx.x * 8 + wave;
    const bool active = (mt < mtiles);
    const int mtc     = active ? mt : (mtiles - 1);
    const int rowbase = mtc * 16;
    const int m  = lane & 15;
    const int kh = lane >> 4;
    const int kb = kh ? 16 : 0;

    // Load resident A operand (16x32 bf16 per k-tile) straight from row-major X:
    // lane holds row m; dwords 0..3 <- K = kb..kb+7, dwords 4..7 <- K = kb+8..kb+15.
    bf16x16 A[KT];
    const unsigned short* xr = X + (size_t)(rowbase + m) * Kdim;
#pragma unroll
    for (int kt = 0; kt < KT; ++kt) {
        bf16x8 lo = *(const bf16x8*)(const void*)(xr + kt * 32 + kb);
        bf16x8 hi = *(const bf16x8*)(const void*)(xr + kt * 32 + kb + 8);
        bf16x16 a;
#pragma unroll
        for (int j = 0; j < 8; ++j) { a[j] = lo[j]; a[8 + j] = hi[j]; }
        A[kt] = a;
    }

    fx8 gacc;
#pragma unroll
    for (int j = 0; j < 8; ++j) gacc[j] = 0.0f;

    const unsigned int* sWc = sW;  // for B-tile reads
    constexpr int NCH = CT / NTCHUNK;
    for (int gch = 0; gch < NCH; ++gch) {
        __syncthreads();
        // Cooperatively stage B chunk: KT*NTCHUNK tiles of 1KB each.
#pragma unroll
        for (int it = 0; it < (KT * NTCHUNK) / 4; ++it) {
            int i4   = tid + 256 * it;
            int sIdx = i4 << 2;
            int p    = sIdx >> 8;  // kt*NTCHUNK + j
            int r    = sIdx & 255;
            int kt   = p / NTCHUNK;
            int j    = p - kt * NTCHUNK;
            *(uint4*)&sW[sIdx] =
                *(const uint4*)&Wp[(((size_t)kt * CT + gch * NTCHUNK + j) << 8) + r];
        }
        __syncthreads();

        // Prefetch next chunk into cache while this chunk computes
        // (lowers to global_prefetch_b8).
        if (gch + 1 < NCH) {
#pragma unroll
            for (int kt = 0; kt < KT; ++kt) {
                if (tid < NTCHUNK * 16) {
                    const char* pc =
                        (const char*)&Wp[(((size_t)kt * CT + (gch + 1) * NTCHUNK) << 8)];
                    __builtin_prefetch(pc + tid * 64, 0, 1);
                }
            }
        }

        for (int ntl = 0; ntl < NTCHUNK; ++ntl) {
            const int tbase = ntl << 8;
            fx8 acc0, acc1;
#pragma unroll
            for (int j = 0; j < 8; ++j) { acc0[j] = 0.0f; acc1[j] = 0.0f; }

            // Software-pipelined over k-tiles (KT is even for all instantiations).
            bf16x16 bA = *(const bf16x16*)(const void*)&sWc[((0 * NTCHUNK) << 8) + tbase + (lane << 3)];
            bf16x16 bB = *(const bf16x16*)(const void*)&sWc[((1 * NTCHUNK) << 8) + tbase + (lane << 3)];
#pragma unroll
            for (int kt = 0; kt < KT; kt += 2) {
                bf16x16 nA = (kt + 2 < KT)
                    ? *(const bf16x16*)(const void*)&sWc[(((kt + 2) * NTCHUNK) << 8) + tbase + (lane << 3)]
                    : bA;
                bf16x16 nB = (kt + 3 < KT)
                    ? *(const bf16x16*)(const void*)&sWc[(((kt + 3) * NTCHUNK) << 8) + tbase + (lane << 3)]
                    : bB;
                acc0 = __builtin_amdgcn_wmma_f32_16x16x32_bf16(
                    false, A[kt], false, bA, (short)0, acc0, false, false);
                acc1 = __builtin_amdgcn_wmma_f32_16x16x32_bf16(
                    false, A[kt + 1], false, bB, (short)0, acc1, false, false);
                bA = nA;
                bB = nB;
            }

            const int col = (gch * NTCHUNK + ntl) * 16 + m;  // n == lane&15
            const float bias = b1[col];
            if constexpr (MODE == 0) {
                if (active) {
#pragma unroll
                    for (int v = 0; v < 8; ++v)
                        Ybf[(size_t)(rowbase + v + 8 * kh) * C + col] =
                            f2bfbits(acc0[v] + acc1[v] + bias);
                }
            } else {
                const float w2c = w2[col];
#pragma unroll
                for (int v = 0; v < 8; ++v) {
                    float t = acc0[v] + acc1[v] + bias;
                    float s = t / (1.0f + __expf(-t));  // silu
                    gacc[v] += s * w2c;
                }
            }
        }
    }

    if constexpr (MODE == 1) {
        // Reduce the 16 column-lanes of each half-wave (they share the same m set).
#pragma unroll
        for (int v = 0; v < 8; ++v) {
            float s = gacc[v];
            s += __shfl_xor(s, 1, 16);
            s += __shfl_xor(s, 2, 16);
            s += __shfl_xor(s, 4, 16);
            s += __shfl_xor(s, 8, 16);
            gacc[v] = s;
        }
        if (((lane & 15) == 0) && active) {
            const float b2v = b2[0];
#pragma unroll
            for (int v = 0; v < 8; ++v)
                Yvec[rowbase + v + 8 * kh] = gacc[v] + b2v;
        }
    }
}

// ---------------------------------------------------------------------------
// Per-graph segment softmax + attention-weighted pooling.
// One block per graph; segment range found by binary search on sorted batch.
// pooled_bf[gid] = sum_i (exp(g_i - max)/denom) * h_i   (bf16 out, 512 cols)
// ---------------------------------------------------------------------------
__global__ __launch_bounds__(256) void pool_kernel(
    const int* __restrict__ batch, const float* __restrict__ g,
    const unsigned short* __restrict__ h_bf,
    unsigned short* __restrict__ pooled_bf, int N) {
    const int gid = blockIdx.x;
    const int tid = threadIdx.x;

    int lo = 0, hi = N;
    while (lo < hi) { int mid = (lo + hi) >> 1; if (batch[mid] < gid) lo = mid + 1; else hi = mid; }
    const int start = lo;
    hi = N;
    while (lo < hi) { int mid = (lo + hi) >> 1; if (batch[mid] < gid + 1) lo = mid + 1; else hi = mid; }
    const int end = lo;

    unsigned int* outp = (unsigned int*)(pooled_bf + (size_t)gid * HH);

    if (start >= end) {          // empty segment -> pooled row = 0
        outp[tid] = 0u;          // 256 dwords == 512 bf16
        return;
    }

    __shared__ float red[256];
    __shared__ float se[64];

    // segment max
    float mx = -3.402823466e38f;
    for (int i = start + tid; i < end; i += 256) mx = fmaxf(mx, g[i]);
    red[tid] = mx;
    __syncthreads();
    for (int s = 128; s > 0; s >>= 1) {
        if (tid < s) red[tid] = fmaxf(red[tid], red[tid + s]);
        __syncthreads();
    }
    mx = red[0];
    __syncthreads();

    // denom = sum exp(g - mx)
    float sm = 0.0f;
    for (int i = start + tid; i < end; i += 256) sm += __expf(g[i] - mx);
    red[tid] = sm;
    __syncthreads();
    for (int s = 128; s > 0; s >>= 1) {
        if (tid < s) red[tid] += red[tid + s];
        __syncthreads();
    }
    const float denom = red[0];
    const float inv   = (denom > 0.0f) ? (1.0f / denom) : 0.0f;

    // weighted sum of h rows; each thread owns cols (2*tid, 2*tid+1)
    float acc0 = 0.0f, acc1 = 0.0f;
    const int c0 = tid * 2;
    for (int base = start; base < end; base += 64) {
        const int cnt = min(64, end - base);
        __syncthreads();
        if (tid < cnt) se[tid] = __expf(g[base + tid] - mx);
        __syncthreads();
        for (int j = 0; j < cnt; ++j) {
            const float e = se[j];
            unsigned int pr =
                *(const unsigned int*)(h_bf + (size_t)(base + j) * HH + c0);
            acc0 += e * bf2f((unsigned short)(pr & 0xffffu));
            acc1 += e * bf2f((unsigned short)(pr >> 16));
        }
    }
    unsigned int o = ((unsigned int)f2bfbits(acc1 * inv) << 16) |
                     (unsigned int)f2bfbits(acc0 * inv);
    outp[tid] = o;
}

// ---------------------------------------------------------------------------
// Launch
// ---------------------------------------------------------------------------
extern "C" void kernel_launch(void* const* d_in, const int* in_sizes, int n_in,
                              void* d_out, int out_size, void* d_ws, size_t ws_size,
                              hipStream_t stream) {
    const float* x        = (const float*)d_in[0];
    const int*   batch    = (const int*)d_in[1];
    // d_in[2] = num_graphs (compile-time GG)
    const float* W_scalar = (const float*)d_in[3];
    const float* b_scalar = (const float*)d_in[4];
    const float* Wg1      = (const float*)d_in[5];
    const float* bg1      = (const float*)d_in[6];
    const float* Wg2      = (const float*)d_in[7];
    const float* bg2      = (const float*)d_in[8];
    const float* Wm1      = (const float*)d_in[9];
    const float* bm1      = (const float*)d_in[10];
    const float* Wm2      = (const float*)d_in[11];
    const float* bm2      = (const float*)d_in[12];
    float* out            = (float*)d_out;

    // Workspace layout (bytes, all 256-aligned): ~318 MB total
    char* ws = (char*)d_ws;
    unsigned short* x_bf      = (unsigned short*)(ws);                 // 102,400,000
    unsigned short* h_bf      = (unsigned short*)(ws + 102400000);     // 204,800,000
    float*          gbuf      = (float*)        (ws + 307200000);     //     800,000
    unsigned int*   Wsp       = (unsigned int*) (ws + 308000000);     //     262,144
    unsigned int*   Wg1p      = (unsigned int*) (ws + 308262144);     //     262,144
    unsigned int*   Wm1p      = (unsigned int*) (ws + 308524288);     //     524,288
    unsigned short* pooled_bf = (unsigned short*)(ws + 309048576);    //   8,388,608

    // 1) pack weights into WMMA B-operand layout (bf16)
    pack_b_kernel<<<256, 256, 0, stream>>>(W_scalar, Wsp, DIN, HH);   // 65536 dwords
    pack_b_kernel<<<256, 256, 0, stream>>>(Wg1, Wg1p, HH, HH / 2);    // 65536 dwords
    pack_b_kernel<<<512, 256, 0, stream>>>(Wm1, Wm1p, HH, HH);        // 131072 dwords

    // 2) x -> bf16 row-major
    const int n4 = (NN * DIN) / 4;  // 12,800,000
    cvt_bf16_kernel<<<n4 / 256, 256, 0, stream>>>((const float4*)x, (uint2*)x_bf, n4);

    // 3) h = x @ W_scalar + b  (bf16 out)   : 12500 M-tiles, 8 waves/block
    mlp_kernel<DIN / 32, HH / 16, 0, 8><<<(12500 + 7) / 8, 256, 0, stream>>>(
        x_bf, Wsp, b_scalar, nullptr, nullptr, h_bf, nullptr, NN / 16);

    // 4) g = silu(h @ Wg1 + bg1) @ Wg2 + bg2
    mlp_kernel<HH / 32, (HH / 2) / 16, 1, 4><<<(12500 + 7) / 8, 256, 0, stream>>>(
        h_bf, Wg1p, bg1, Wg2, bg2, nullptr, gbuf, NN / 16);

    // 5) segment softmax + attention pooling -> pooled (bf16, G x 512)
    pool_kernel<<<GG, 256, 0, stream>>>(batch, gbuf, h_bf, pooled_bf, NN);

    // 6) out = silu(pooled @ Wm1 + bm1) @ Wm2 + bm2
    mlp_kernel<HH / 32, HH / 16, 1, 4><<<(GG / 16 + 7) / 8, 256, 0, stream>>>(
        pooled_bf, Wm1p, bm1, Wm2, bm2, nullptr, out, GG / 16);
}
